// SnakeBlock_89421219103310
// MI455X (gfx1250) — compile-verified, matrix operand order checked by reading, not run on hardware
//
#include <hip/hip_runtime.h>

typedef _Float16 v16h __attribute__((ext_vector_type(16)));
typedef _Float16 v8h  __attribute__((ext_vector_type(8)));
typedef float    v8f  __attribute__((ext_vector_type(8)));

#define B_IMG   8
#define C_DIM   256
#define H_DIM   128
#define W_DIM   128
#define EPI     4096
#define N_PTS   (B_IMG * EPI * 2)   /* 65536 sample points */

// ---------------------------------------------------------------------------
// Kernel 0: convert W1 (C x C, row-major [j][k]) fp32 -> f16
// ---------------------------------------------------------------------------
__global__ void convert_w1_kernel(const float* __restrict__ W1,
                                  _Float16* __restrict__ W1h) {
    int i = blockIdx.x * blockDim.x + threadIdx.x;
    if (i < C_DIM * C_DIM) W1h[i] = (_Float16)W1[i];
}

// ---------------------------------------------------------------------------
// Kernel 1: bilinear gather.  One wave per point; lane c handles channels
// c, c+32, ... .  Writes sampled activations as f16, row-major (N_PTS x C).
// ---------------------------------------------------------------------------
__global__ void __launch_bounds__(256)
sample_kernel(const float* __restrict__ features,
              const float* __restrict__ edges,
              _Float16* __restrict__ A) {
    int gwave = (blockIdx.x * blockDim.x + threadIdx.x) >> 5;
    int lane  = threadIdx.x & 31;
    int n = gwave;
    if (n >= N_PTS) return;

    int bid = n >> 13;                 // n / (2*EPI) = n / 8192
    float ptx = edges[2 * n + 0];
    float pty = edges[2 * n + 1];

    // px = ((pt*2/H - 1) + 1) * W/2 - 0.5 ; same for py with H
    const float sx = (2.0f / (float)H_DIM) * ((float)W_DIM * 0.5f);
    const float sy = (2.0f / (float)H_DIM) * ((float)H_DIM * 0.5f);
    float px = ptx * sx - 0.5f;
    float py = pty * sy - 0.5f;

    float x0f = floorf(px), y0f = floorf(py);
    int x0 = (int)x0f, y0 = (int)y0f;
    int x1 = x0 + 1,   y1 = y0 + 1;
    float wx1 = px - x0f, wy1 = py - y0f;
    float wx0 = 1.0f - wx1, wy0 = 1.0f - wy1;

    float mx0 = (x0 >= 0 && x0 < W_DIM) ? 1.0f : 0.0f;
    float mx1 = (x1 >= 0 && x1 < W_DIM) ? 1.0f : 0.0f;
    float my0 = (y0 >= 0 && y0 < H_DIM) ? 1.0f : 0.0f;
    float my1 = (y1 >= 0 && y1 < H_DIM) ? 1.0f : 0.0f;

    int x0c = min(max(x0, 0), W_DIM - 1);
    int x1c = min(max(x1, 0), W_DIM - 1);
    int y0c = min(max(y0, 0), H_DIM - 1);
    int y1c = min(max(y1, 0), H_DIM - 1);

    float w00 = wx0 * wy0 * mx0 * my0;
    float w10 = wx1 * wy0 * mx1 * my0;
    float w01 = wx0 * wy1 * mx0 * my1;
    float w11 = wx1 * wy1 * mx1 * my1;

    int o00 = y0c * W_DIM + x0c;
    int o10 = y0c * W_DIM + x1c;
    int o01 = y1c * W_DIM + x0c;
    int o11 = y1c * W_DIM + x1c;

    const float* fb = features + (size_t)bid * C_DIM * H_DIM * W_DIM;
    _Float16* arow = A + (size_t)n * C_DIM;

    #pragma unroll
    for (int i = 0; i < C_DIM / 32; ++i) {
        int c = lane + 32 * i;
        const float* fc = fb + (size_t)c * (H_DIM * W_DIM);
        float v = w00 * fc[o00] + w10 * fc[o10] + w01 * fc[o01] + w11 * fc[o11];
        arow[c] = (_Float16)v;
    }
}

// ---------------------------------------------------------------------------
// Kernel 2: fused WMMA GEMM h = relu(A @ W1^T + b1), d = h @ W2^T,
//           out = edges + d.   One 16-row tile per wave, 8 waves / block.
// ---------------------------------------------------------------------------
__global__ void __launch_bounds__(256)
gemm_kernel(const _Float16* __restrict__ A,
            const _Float16* __restrict__ W1h,
            const float* __restrict__ b1,
            const float* __restrict__ W2,
            const float* __restrict__ edges,
            float* __restrict__ out) {
    int lane = threadIdx.x & 31;
    int wave = threadIdx.x >> 5;
    int rowBase = (blockIdx.x * 8 + wave) * 16;
    int half = lane >> 4;           // 0: lanes 0-15, 1: lanes 16-31
    int l16  = lane & 15;

    // Preload all 8 A-fragments (K = 0..255) for this 16-row tile.
    // ISA f16 A layout: elements 0..7  <-> K = k0 + 8*half + (0..7)
    //                   elements 8..15 <-> K = k0 + 16 + 8*half + (0..7)
    const _Float16* arow = A + (size_t)(rowBase + l16) * C_DIM;
    v16h afrag[8];
    #pragma unroll
    for (int s = 0; s < 8; ++s) {
        int k0 = 32 * s;
        v8h lo = *(const v8h*)(arow + k0 + half * 8);
        v8h hi = *(const v8h*)(arow + k0 + 16 + half * 8);
        afrag[s] = __builtin_shufflevector(lo, hi,
                     0, 1, 2, 3, 4, 5, 6, 7, 8, 9, 10, 11, 12, 13, 14, 15);
    }

    v8f d0 = {};   // per-lane partial sums of h * W2[0][col]
    v8f d1 = {};   // per-lane partial sums of h * W2[1][col]

    for (int t = 0; t < 16; ++t) {           // 16 column tiles of 16
        int col = t * 16 + l16;
        const _Float16* brow = W1h + (size_t)col * C_DIM;  // W1^T column = W1 row
        v8f acc = {};
        #pragma unroll
        for (int s = 0; s < 8; ++s) {
            int k0 = 32 * s;
            v8h lo = *(const v8h*)(brow + k0 + half * 8);
            v8h hi = *(const v8h*)(brow + k0 + 16 + half * 8);
            v16h bfrag = __builtin_shufflevector(lo, hi,
                           0, 1, 2, 3, 4, 5, 6, 7, 8, 9, 10, 11, 12, 13, 14, 15);
            acc = __builtin_amdgcn_wmma_f32_16x16x32_f16(
                      false, afrag[s], false, bfrag,
                      (short)0, acc, false, false);
        }
        // Epilogue for this tile: bias + relu, project through W2.
        float bias = b1[col];
        float w20  = W2[col];
        float w21  = W2[C_DIM + col];
        #pragma unroll
        for (int v = 0; v < 8; ++v) {
            float h = acc[v] + bias;         // row M = v + 8*half, col = col
            h = h > 0.0f ? h : 0.0f;
            d0[v] += h * w20;
            d1[v] += h * w21;
        }
    }

    // Reduce the column-partials across the 16 lanes of each half-group.
    #pragma unroll
    for (int off = 1; off < 16; off <<= 1) {
        #pragma unroll
        for (int v = 0; v < 8; ++v) {
            d0[v] += __shfl_xor(d0[v], off, 32);
            d1[v] += __shfl_xor(d1[v], off, 32);
        }
    }

    // Lane 0 writes rows 0..7, lane 16 writes rows 8..15 of the tile.
    if (l16 == 0) {
        #pragma unroll
        for (int v = 0; v < 8; ++v) {
            int n = rowBase + half * 8 + v;
            out[2 * n + 0] = edges[2 * n + 0] + d0[v];
            out[2 * n + 1] = edges[2 * n + 1] + d1[v];
        }
    }
}

// ---------------------------------------------------------------------------
extern "C" void kernel_launch(void* const* d_in, const int* in_sizes, int n_in,
                              void* d_out, int out_size, void* d_ws, size_t ws_size,
                              hipStream_t stream) {
    const float* features = (const float*)d_in[0];   // (8,256,128,128) f32
    const float* edges    = (const float*)d_in[1];   // (32768,4) f32
    const float* W1       = (const float*)d_in[2];   // (256,256) f32
    const float* b1       = (const float*)d_in[3];   // (256,)   f32
    const float* W2       = (const float*)d_in[4];   // (2,256)  f32
    // d_in[5] = edges_per_img scalar (compile-time constant here)

    _Float16* W1h  = (_Float16*)d_ws;                                 // 128 KB
    _Float16* Abuf = (_Float16*)((char*)d_ws +
                                 (size_t)C_DIM * C_DIM * sizeof(_Float16)); // 32 MB

    convert_w1_kernel<<<(C_DIM * C_DIM + 255) / 256, 256, 0, stream>>>(W1, W1h);
    sample_kernel<<<N_PTS / 8, 256, 0, stream>>>(features, edges, Abuf);
    gemm_kernel<<<N_PTS / (16 * 8), 256, 0, stream>>>(Abuf, W1h, b1, W2, edges,
                                                      (float*)d_out);
}
